// DAT_55216099558171
// MI455X (gfx1250) — matrix operand order repaired and checked by reading, not compiled, not run on hardware
//
#include <hip/hip_runtime.h>
#include <hip/hip_bf16.h>
#include <math.h>
#include <stdint.h>

typedef __attribute__((ext_vector_type(16))) _Float16 v16h;
typedef __attribute__((ext_vector_type(8)))  _Float16 v8h;
typedef __attribute__((ext_vector_type(8)))  float    v8f;
typedef int v4i __attribute__((__vector_size__(16)));

union AF { v16h v; v8h p[2]; _Float16 h[16]; };
union CF { v8f  v; float    f[8];  };
union H8 { v8h  v; _Float16 h[8];  };

constexpr int kC  = 512;
constexpr int kHW = 4096;   // 64*64 queries
constexpr int kNS = 1024;   // 32*32 sampled keys

// ---- CDNA5 async global->LDS path (guarded; falls back to plain copies) ----
#if defined(__has_builtin)
#if __has_builtin(__builtin_amdgcn_global_load_async_to_lds_b128)
#define HAVE_ASYNC_LDS 1
#endif
#endif
#ifndef HAVE_ASYNC_LDS
#define HAVE_ASYNC_LDS 0
#endif

#if HAVE_ASYNC_LDS
// builtin signature (from compiler diagnostic): param0 = v4i __device__* (AS1)
__device__ __forceinline__ __attribute__((address_space(1))) v4i* as_g4(const void* p) {
  return (__attribute__((address_space(1))) v4i*)(uintptr_t)p;
}
__device__ __forceinline__ __attribute__((address_space(3))) v4i* as_l4(void* p) {
  // generic LDS pointer carries the LDS byte offset in its low 32 bits
  return (__attribute__((address_space(3))) v4i*)(uintptr_t)(unsigned)(uintptr_t)p;
}
#endif

// ------------------------------------------------------------------
// elementwise f32 -> f16
// ------------------------------------------------------------------
__global__ void cvt_f16_kernel(const float* __restrict__ s, _Float16* __restrict__ d, int n) {
  int i = blockIdx.x * 256 + threadIdx.x;
  if (i < n) d[i] = (_Float16)s[i];
}

// ------------------------------------------------------------------
// WMMA GEMM: C[M,N] = A[M,K]*B[K,N] + bias[M]; A,B f16 row-major.
// Optional f32 output Cf, optional f16 output Ch (row-major or transposed).
// Block tile 64x128, 8 waves, wave tile 32x32 (2x2 WMMA), f32 accumulate.
// All fragment LDS reads are contiguous 16B/32B vectors.
// ------------------------------------------------------------------
__global__ void __launch_bounds__(256)
gemm_wmma_f16(const _Float16* __restrict__ A, const _Float16* __restrict__ Bm,
              const float* __restrict__ bias, float* __restrict__ Cf,
              _Float16* __restrict__ Ch, int M, int N, int K, int transCh)
{
  __shared__ _Float16 As [64][32];    // [m][k]
  __shared__ _Float16 BsT[128][32];   // [n][k]  (transposed B tile)
  const int tid  = threadIdx.x;
  const int lane = tid & 31, wave = tid >> 5;
  const int bm = blockIdx.y, bn = blockIdx.x;
  const int wm = wave >> 2, wn = wave & 3;
  const int lr = lane & 15, hi = lane >> 4;
  const int kb = hi * 8, kb2 = hi * 16;

  CF acc[2][2] = {};

  for (int k0 = 0; k0 < K; k0 += 32) {
    __syncthreads();
    {  // A tile 64x32, vector copy (1 x b128 per thread)
      int r = tid >> 2, c0 = (tid & 3) * 8;
      *(v8h*)&As[r][c0] = *(const v8h*)(A + (bm * 64 + r) * K + k0 + c0);
    }
#pragma unroll
    for (int i = 0; i < 2; ++i) {  // B tile 32x128 -> BsT (vector global read)
      int idx = tid + i * 256;
      int rr = idx >> 4, c0 = (idx & 15) * 8;
      H8 t; t.v = *(const v8h*)(Bm + (k0 + rr) * N + bn * 128 + c0);
#pragma unroll
      for (int j = 0; j < 8; ++j) BsT[c0 + j][rr] = t.h[j];
    }
    __syncthreads();

    AF a[2], b[2];
#pragma unroll
    for (int sm = 0; sm < 2; ++sm) {
      int row = wm * 32 + sm * 16 + lr;
      a[sm].p[0] = *(const v8h*)&As[row][kb2];
      a[sm].p[1] = *(const v8h*)&As[row][kb2 + 8];
    }
#pragma unroll
    for (int sn = 0; sn < 2; ++sn) {
      int col = wn * 32 + sn * 16 + lr;
      b[sn].v = *(const v16h*)&BsT[col][kb2];
    }
#pragma unroll
    for (int sm = 0; sm < 2; ++sm)
#pragma unroll
      for (int sn = 0; sn < 2; ++sn)
        acc[sm][sn].v = __builtin_amdgcn_wmma_f32_16x16x32_f16(
            false, a[sm].v, false, b[sn].v, (short)0, acc[sm][sn].v, false, false);
  }

  // -------- epilogue: hoisted branches, pure 32-bit index math --------
  float bv[2][8];
  if (bias) {
#pragma unroll
    for (int sm = 0; sm < 2; ++sm)
#pragma unroll
      for (int r = 0; r < 8; ++r)
        bv[sm][r] = bias[bm * 64 + wm * 32 + sm * 16 + hi * 8 + r];
  } else {
#pragma unroll
    for (int sm = 0; sm < 2; ++sm)
#pragma unroll
      for (int r = 0; r < 8; ++r) bv[sm][r] = 0.0f;
  }

  if (Cf) {
#pragma unroll
    for (int sm = 0; sm < 2; ++sm)
#pragma unroll
      for (int sn = 0; sn < 2; ++sn)
#pragma unroll
        for (int r = 0; r < 8; ++r) {
          int row = bm * 64 + wm * 32 + sm * 16 + hi * 8 + r;
          int col = bn * 128 + wn * 32 + sn * 16 + lr;
          Cf[row * N + col] = acc[sm][sn].f[r] + bv[sm][r];
        }
  }
  if (Ch) {
    if (transCh) {
#pragma unroll
      for (int sm = 0; sm < 2; ++sm)
#pragma unroll
        for (int sn = 0; sn < 2; ++sn)
#pragma unroll
          for (int r = 0; r < 8; ++r) {
            int row = bm * 64 + wm * 32 + sm * 16 + hi * 8 + r;
            int col = bn * 128 + wn * 32 + sn * 16 + lr;
            Ch[col * M + row] = (_Float16)(acc[sm][sn].f[r] + bv[sm][r]);
          }
    } else {
#pragma unroll
      for (int sm = 0; sm < 2; ++sm)
#pragma unroll
        for (int sn = 0; sn < 2; ++sn)
#pragma unroll
          for (int r = 0; r < 8; ++r) {
            int row = bm * 64 + wm * 32 + sm * 16 + hi * 8 + r;
            int col = bn * 128 + wn * 32 + sn * 16 + lr;
            Ch[row * N + col] = (_Float16)(acc[sm][sn].f[r] + bv[sm][r]);
          }
    }
  }
}

// ------------------------------------------------------------------
// Offset network: depthwise 5x5 s2 conv -> LN(channels) -> GELU -> 1x1(2ch)
// -> tanh * 2/32 + ref. One block per (group, oy, ox); 128 threads = channels.
// ------------------------------------------------------------------
__device__ __forceinline__ float block_sum_128(float v, float* red, int c) {
  red[c] = v; __syncthreads();
#pragma unroll
  for (int s = 64; s > 0; s >>= 1) {
    if (c < s) red[c] += red[c + s];
    __syncthreads();
  }
  float r = red[0];
  __syncthreads();
  return r;
}

__global__ void __launch_bounds__(128)
offset_net(const float* __restrict__ q, const float* __restrict__ wdw,
           const float* __restrict__ bdw, const float* __restrict__ lng,
           const float* __restrict__ lnb, const float* __restrict__ wpw,
           float* __restrict__ pos_out, float* __restrict__ ref_out)
{
  __shared__ float red[128];
  const int c   = threadIdx.x;
  const int bid = blockIdx.x;                    // g*1024 + oy*32 + ox
  const int g = bid >> 10, oy = (bid >> 5) & 31, ox = bid & 31;
  const float* plane = q + (g * 128 + c) * 4096;

  float t = bdw[c];
  const int iy0 = oy * 2 - 2, ix0 = ox * 2 - 2;
#pragma unroll
  for (int ky = 0; ky < 5; ++ky) {
    int iy = iy0 + ky;
    if (iy < 0 || iy > 63) continue;
#pragma unroll
    for (int kx = 0; kx < 5; ++kx) {
      int ix = ix0 + kx;
      if (ix < 0 || ix > 63) continue;
      t += wdw[c * 25 + ky * 5 + kx] * plane[iy * 64 + ix];
    }
  }
  float mu  = block_sum_128(t, red, c) * (1.0f / 128.0f);
  float d   = t - mu;
  float var = block_sum_128(d * d, red, c) * (1.0f / 128.0f);
  float tn  = d * rsqrtf(var + 1e-5f) * lng[c] + lnb[c];
  float ge  = 0.5f * tn * (1.0f + erff(tn * 0.70710678f));
  float s0  = block_sum_128(wpw[c] * ge, red, c);        // y channel
  float s1  = block_sum_128(wpw[128 + c] * ge, red, c);  // x channel
  if (c == 0) {
    float offy = tanhf(s0) * (2.0f / 32.0f);
    float offx = tanhf(s1) * (2.0f / 32.0f);
    float refy = (oy + 0.5f) * (2.0f / 32.0f) - 1.0f;
    float refx = (ox + 0.5f) * (2.0f / 32.0f) - 1.0f;
    pos_out[bid * 2 + 0] = offy + refy;
    pos_out[bid * 2 + 1] = offx + refx;
    ref_out[bid * 2 + 0] = refy;
    ref_out[bid * 2 + 1] = refx;
  }
}

// ------------------------------------------------------------------
// Bilinear sample of x (grouped) at pos -> xs f16 [512][1024]
// ------------------------------------------------------------------
__global__ void __launch_bounds__(128)
sample_xs(const float* __restrict__ x, const float* __restrict__ posb,
          _Float16* __restrict__ xsh)
{
  const int bid = blockIdx.x;                    // g*1024 + n
  const int g = bid >> 10, n = bid & 1023;
  const int c = threadIdx.x;
  float py = posb[bid * 2 + 0];
  float px = posb[bid * 2 + 1];
  float gx = (px + 1.0f) * 0.5f * 63.0f;
  float gy = (py + 1.0f) * 0.5f * 63.0f;
  float x0f = floorf(gx), y0f = floorf(gy);
  float wx = gx - x0f, wy = gy - y0f;
  int ix = (int)x0f, iy = (int)y0f;
  const float* plane = x + (g * 128 + c) * 4096;
  auto tap = [&](int xx, int yy) -> float {
    return (xx >= 0 && xx < 64 && yy >= 0 && yy < 64) ? plane[yy * 64 + xx] : 0.0f;
  };
  float v = tap(ix, iy) * (1 - wx) * (1 - wy) + tap(ix + 1, iy) * wx * (1 - wy)
          + tap(ix, iy + 1) * (1 - wx) * wy   + tap(ix + 1, iy + 1) * wx * wy;
  xsh[(g * 128 + c) * kNS + n] = (_Float16)v;
}

// ------------------------------------------------------------------
// Fused attention per head.
// khT is k stored transposed [ns=1024][C=512] f16 so B fragments are direct
// aligned global b128 loads. LDS: S [32][1024] f32 (128KB) + Q^T [32][64] f16
// (4KB) + rpe head slice [127*127] f32 (64.5KB) -> ~195KB (CDNA5 WGP LDS).
// ------------------------------------------------------------------
#define SMEM_ATTN (32 * 1024 * 4 + 32 * 64 * 2 + 16129 * 4)

__device__ __forceinline__ float rpe_tap(const float* t, int ix, int iy) {
  return (ix >= 0 && ix < 127 && iy >= 0 && iy < 127) ? t[iy * 127 + ix] : 0.0f;
}

__global__ void __launch_bounds__(256)
attn_fused(const _Float16* __restrict__ qh, const _Float16* __restrict__ khT,
           const _Float16* __restrict__ vh, const float* __restrict__ posb,
           const float* __restrict__ rpe, _Float16* __restrict__ outT)
{
  extern __shared__ char smem[];
  float*    Sb = (float*)smem;                            // [32][1024]
  _Float16* Qs = (_Float16*)(smem + 32 * 1024 * 4);       // [32][64] (m-major)
  float*    Rp = (float*)(smem + 32 * 1024 * 4 + 32 * 64 * 2); // [16129]

  const int tid  = threadIdx.x;
  const int lane = tid & 31, wave = tid >> 5;
  const int lr = lane & 15, hi = lane >> 4;
  const int kb = hi * 8, kb2 = hi * 16;
  const int h = blockIdx.y;
  const int mbase = blockIdx.x * 32;
  const int g = h >> 1;
  const float scale = 0.125f;                 // 64^-0.5
  const float* rpeh = rpe + h * 16129;

  // ---- stage rpe head slice into LDS (async copy when available) ----
#if HAVE_ASYNC_LDS
  for (int i = tid; i < 4032; i += 256)       // 4032*16B = 16128 floats
    __builtin_amdgcn_global_load_async_to_lds_b128(
        as_g4(rpeh + i * 4), as_l4(Rp + i * 4), 0, 0);
  if (tid == 0) Rp[16128] = rpeh[16128];
#if __has_builtin(__builtin_amdgcn_s_wait_asynccnt)
  __builtin_amdgcn_s_wait_asynccnt(0);
#else
  asm volatile("s_wait_asynccnt 0" ::: "memory");
#endif
#else
  for (int i = tid; i < 16129; i += 256) Rp[i] = rpeh[i];
#endif

  // ---- stage Q^T block transposed: Qs[m][c] ----
  {
    int c = tid >> 2, mc = (tid & 3) * 8;
    H8 t; t.v = *(const v8h*)(qh + (h * 64 + c) * kHW + mbase + mc);
#pragma unroll
    for (int j = 0; j < 8; ++j) Qs[(mc + j) * 64 + c] = t.h[j];
  }
  __syncthreads();

  // ---- S = scale * Q^T K ----
  for (int t = wave; t < 128; t += 8) {
    int tm = t >> 6, tn = t & 63;
    CF acc = {};
#pragma unroll
    for (int ks = 0; ks < 64; ks += 32) {
      AF a, b;
      int m = tm * 16 + lr;
      a.p[0] = *(const v8h*)(Qs + m * 64 + ks + kb2);
      a.p[1] = *(const v8h*)(Qs + m * 64 + ks + kb2 + 8);
      b.v    = *(const v16h*)(khT + (tn * 16 + lr) * kC + h * 64 + ks + kb2);
      acc.v = __builtin_amdgcn_wmma_f32_16x16x32_f16(false, a.v, false, b.v,
                                                     (short)0, acc.v, false, false);
    }
#pragma unroll
    for (int r = 0; r < 8; ++r)
      Sb[(tm * 16 + hi * 8 + r) * 1024 + tn * 16 + lr] = acc.f[r] * scale;
  }
  __syncthreads();

  // ---- add relative-position bias from LDS-resident rpe slice ----
  for (int i = 0; i < 128; ++i) {
    int idx = tid + i * 256;                  // 32*1024 elements
    int m = idx >> 10, n = idx & 1023;
    int mg = mbase + m;
    float qy = ((mg >> 6) + 0.5f) * (2.0f / 64.0f) - 1.0f;
    float qx = ((mg & 63) + 0.5f) * (2.0f / 64.0f) - 1.0f;
    float py = posb[(g * 1024 + n) * 2 + 0];
    float px = posb[(g * 1024 + n) * 2 + 1];
    float gx = ((qx - px) * 0.5f + 1.0f) * 0.5f * 126.0f;
    float gy = ((qy - py) * 0.5f + 1.0f) * 0.5f * 126.0f;
    float x0f = floorf(gx), y0f = floorf(gy);
    float wx = gx - x0f, wy = gy - y0f;
    int ix = (int)x0f, iy = (int)y0f;
    float bias = rpe_tap(Rp, ix, iy) * (1 - wx) * (1 - wy)
               + rpe_tap(Rp, ix + 1, iy) * wx * (1 - wy)
               + rpe_tap(Rp, ix, iy + 1) * (1 - wx) * wy
               + rpe_tap(Rp, ix + 1, iy + 1) * wx * wy;
    Sb[idx] += bias;
  }
  __syncthreads();

  // ---- softmax over keys: 4 rows per wave, wave32 shfl reductions ----
#pragma unroll
  for (int rr = 0; rr < 4; ++rr) {
    float* row = Sb + (wave * 4 + rr) * 1024;
    float mx = -3.0e38f;
    for (int n = lane; n < 1024; n += 32) mx = fmaxf(mx, row[n]);
#pragma unroll
    for (int o = 16; o > 0; o >>= 1) mx = fmaxf(mx, __shfl_xor(mx, o, 32));
    float sum = 0.0f;
    for (int n = lane; n < 1024; n += 32) {
      float e = __expf(row[n] - mx);
      row[n] = e; sum += e;
    }
#pragma unroll
    for (int o = 16; o > 0; o >>= 1) sum += __shfl_xor(sum, o, 32);
    float inv = 1.0f / sum;
    for (int n = lane; n < 1024; n += 32) row[n] *= inv;
  }
  __syncthreads();

  // ---- O^T[m][c] = P * V^T ; one 16x16 tile per wave ----
  {
    int tm = wave >> 2, tc = wave & 3;
    CF acc = {};
    for (int ks = 0; ks < 1024; ks += 32) {
      AF a, b;
      const float* srow = Sb + (tm * 16 + lr) * 1024 + ks;
      CF t0, t1;
      t0.v = *(const v8f*)(srow + kb2);
      t1.v = *(const v8f*)(srow + kb2 + 8);
#pragma unroll
      for (int e = 0; e < 8; ++e) {
        a.h[e]     = (_Float16)t0.f[e];
        a.h[8 + e] = (_Float16)t1.f[e];
      }
      b.v = *(const v16h*)(vh + (h * 64 + tc * 16 + lr) * kNS + ks + kb2);
      acc.v = __builtin_amdgcn_wmma_f32_16x16x32_f16(false, a.v, false, b.v,
                                                     (short)0, acc.v, false, false);
    }
    H8 o;
#pragma unroll
    for (int r = 0; r < 8; ++r) o.h[r] = (_Float16)acc.f[r];
    *(v8h*)(outT + (h * 64 + tc * 16 + lr) * kHW + mbase + tm * 16 + hi * 8) = o.v;
  }
}

// ------------------------------------------------------------------
extern "C" void kernel_launch(void* const* d_in, const int* in_sizes, int n_in,
                              void* d_out, int out_size, void* d_ws, size_t ws_size,
                              hipStream_t stream) {
  (void)in_sizes; (void)n_in; (void)out_size; (void)ws_size;
  const float* x   = (const float*)d_in[0];
  const float* wq  = (const float*)d_in[1];
  const float* bq  = (const float*)d_in[2];
  const float* wk  = (const float*)d_in[3];
  const float* bk  = (const float*)d_in[4];
  const float* wv  = (const float*)d_in[5];
  const float* bv  = (const float*)d_in[6];
  const float* wo  = (const float*)d_in[7];
  const float* bo  = (const float*)d_in[8];
  const float* wdw = (const float*)d_in[9];
  const float* bdw = (const float*)d_in[10];
  const float* lng = (const float*)d_in[11];
  const float* lnb = (const float*)d_in[12];
  const float* wpw = (const float*)d_in[13];
  const float* rpe = (const float*)d_in[14];

  char* ws = (char*)d_ws;
  size_t o = 0;
  _Float16* xh  = (_Float16*)(ws + o); o += (size_t)kC * kHW * 2;   // 4 MB
  _Float16* wqh = (_Float16*)(ws + o); o += (size_t)kC * kC * 2;
  _Float16* wkh = (_Float16*)(ws + o); o += (size_t)kC * kC * 2;
  _Float16* wvh = (_Float16*)(ws + o); o += (size_t)kC * kC * 2;
  _Float16* woh = (_Float16*)(ws + o); o += (size_t)kC * kC * 2;
  float*    qf  = (float*)(ws + o);    o += (size_t)kC * kHW * 4;   // 8 MB
  _Float16* qh  = (_Float16*)(ws + o); o += (size_t)kC * kHW * 2;
  _Float16* xsh = (_Float16*)(ws + o); o += (size_t)kC * kNS * 2;
  _Float16* khT = (_Float16*)(ws + o); o += (size_t)kC * kNS * 2;   // [ns][C]
  _Float16* vhb = (_Float16*)(ws + o); o += (size_t)kC * kNS * 2;   // [C][ns]
  _Float16* otT = (_Float16*)(ws + o); o += (size_t)kC * kHW * 2;

  float* yout    = (float*)d_out;
  float* pos_out = yout + (size_t)kC * kHW;   // [1,4,32,32,2]
  float* ref_out = pos_out + 2048;            // [1,4,32,32,2]

  // f32 -> f16 conversions
  cvt_f16_kernel<<<dim3(8192), 256, 0, stream>>>(x,  xh,  kC * kHW);
  cvt_f16_kernel<<<dim3(1024), 256, 0, stream>>>(wq, wqh, kC * kC);
  cvt_f16_kernel<<<dim3(1024), 256, 0, stream>>>(wk, wkh, kC * kC);
  cvt_f16_kernel<<<dim3(1024), 256, 0, stream>>>(wv, wvh, kC * kC);
  cvt_f16_kernel<<<dim3(1024), 256, 0, stream>>>(wo, woh, kC * kC);

  // q = wq*x + bq : f32 (offset net) + f16 (attention)
  gemm_wmma_f16<<<dim3(32, 8), 256, 0, stream>>>(wqh, xh, bq, qf, qh, kC, kHW, kC, 0);

  // offset network -> pos, ref
  offset_net<<<dim3(4096), 128, 0, stream>>>(qf, wdw, bdw, lng, lnb, wpw, pos_out, ref_out);

  // deformable sampling -> xs f16 [512][1024]
  sample_xs<<<dim3(4096), 128, 0, stream>>>(x, pos_out, xsh);

  // k = wk*xs + bk stored TRANSPOSED [ns][C]; v = wv*xs + bv stored [C][ns]
  gemm_wmma_f16<<<dim3(8, 8), 256, 0, stream>>>(wkh, xsh, bk, nullptr, khT, kC, kNS, kC, 1);
  gemm_wmma_f16<<<dim3(8, 8), 256, 0, stream>>>(wvh, xsh, bv, nullptr, vhb, kC, kNS, kC, 0);

  // fused attention -> O^T f16 [C][HW]
  attn_fused<<<dim3(128, 8), 256, SMEM_ATTN, stream>>>(qh, khT, vhb, pos_out, rpe, otT);

  // y = wo*out + bo -> d_out f32 [512][4096]
  gemm_wmma_f16<<<dim3(32, 8), 256, 0, stream>>>(woh, otT, bo, yout, nullptr, kC, kHW, kC, 0);
}
